// cylinder_fea_78795470012490
// MI455X (gfx1250) — matrix operand order repaired and checked by reading, not compiled, not run on hardware
//
#include <hip/hip_runtime.h>
#include <hip/hip_bf16.h>
#include <math.h>

typedef __attribute__((ext_vector_type(16))) __bf16 v16bf;
typedef __attribute__((ext_vector_type(8)))  __bf16 v8bf;
typedef __attribute__((ext_vector_type(8)))  float  v8f;

// CDNA5 async global->LDS path (ASYNCcnt-tracked, no VGPR round trip), guarded
// so the kernel still compiles on toolchains without the builtins.
#if defined(__gfx1250__) && __has_builtin(__builtin_amdgcn_global_load_async_to_lds_b128) && \
    __has_builtin(__builtin_amdgcn_s_wait_asynccnt)
#define USE_ASYNC_LDS 1
// builtin signature (from hipcc diagnostic): (v4i __device__*, v4i LDS*, imm, imm)
typedef int v4i __attribute__((vector_size(16)));
typedef __attribute__((address_space(1))) v4i g_v4i;   // global / device AS
typedef __attribute__((address_space(3))) v4i l_v4i;   // LDS AS
#else
#define USE_ASYNC_LDS 0
#endif

// ---------------- workspace layout (bytes) ----------------
// w1t bf16 [64][32]   @      0  (4096)
// w2t bf16 [128][64]  @   4096  (16384)
// w3t bf16 [256][128] @  20480  (65536)
// w4t bf16 [256][256] @  86016  (131072)
// scales  float[1170] @ 217088  (s0[9] t0[9] s1[64] t1[64] s2[128] t2[128] s3[256] t3[256] b4[256])
#define OFF_W2T 4096
#define OFF_W3T 20480
#define OFF_W4T 86016
#define OFF_SC  217088

// ---------------- prep: fold BN, transpose weights to bf16 ----------------
__global__ void prep_kernel(
    const float* g0, const float* be0, const float* m0, const float* va0,
    const float* g1, const float* be1, const float* m1, const float* va1,
    const float* g2, const float* be2, const float* m2, const float* va2,
    const float* g3, const float* be3, const float* m3, const float* va3,
    const float* w1, const float* b1, const float* w2, const float* b2,
    const float* w3, const float* b3, const float* w4, const float* b4,
    char* ws)
{
  __bf16* w1t = (__bf16*)(ws);
  __bf16* w2t = (__bf16*)(ws + OFF_W2T);
  __bf16* w3t = (__bf16*)(ws + OFF_W3T);
  __bf16* w4t = (__bf16*)(ws + OFF_W4T);
  float*  SC  = (float*)(ws + OFF_SC);
  const int NW1 = 2048, NW2 = 8192, NW3 = 32768, NW4 = 65536;
  const int TOT = NW1 + NW2 + NW3 + NW4 + 1170;
  int stride = gridDim.x * blockDim.x;
  for (int idx = blockIdx.x * blockDim.x + threadIdx.x; idx < TOT; idx += stride) {
    if (idx < NW1) {
      int o = idx >> 5, k = idx & 31;
      w1t[idx] = (k < 9) ? (__bf16)w1[k * 64 + o] : (__bf16)0.f;
    } else if (idx < NW1 + NW2) {
      int j = idx - NW1; int o = j >> 6, k = j & 63;
      w2t[j] = (__bf16)w2[k * 128 + o];
    } else if (idx < NW1 + NW2 + NW3) {
      int j = idx - NW1 - NW2; int o = j >> 7, k = j & 127;
      w3t[j] = (__bf16)w3[k * 256 + o];
    } else if (idx < NW1 + NW2 + NW3 + NW4) {
      int j = idx - NW1 - NW2 - NW3; int o = j >> 8, k = j & 255;
      w4t[j] = (__bf16)w4[k * 256 + o];
    } else {
      int j = idx - (NW1 + NW2 + NW3 + NW4);
      if (j < 9)        { SC[j] = g0[j] * rsqrtf(va0[j] + 1e-5f); }
      else if (j < 18)  { int c = j - 9;   float s = g0[c] * rsqrtf(va0[c] + 1e-5f); SC[j] = be0[c] - m0[c] * s; }
      else if (j < 82)  { int c = j - 18;  SC[j] = g1[c] * rsqrtf(va1[c] + 1e-5f); }
      else if (j < 146) { int c = j - 82;  float s = g1[c] * rsqrtf(va1[c] + 1e-5f); SC[j] = (b1[c] - m1[c]) * s + be1[c]; }
      else if (j < 274) { int c = j - 146; SC[j] = g2[c] * rsqrtf(va2[c] + 1e-5f); }
      else if (j < 402) { int c = j - 274; float s = g2[c] * rsqrtf(va2[c] + 1e-5f); SC[j] = (b2[c] - m2[c]) * s + be2[c]; }
      else if (j < 658) { int c = j - 402; SC[j] = g3[c] * rsqrtf(va3[c] + 1e-5f); }
      else if (j < 914) { int c = j - 658; float s = g3[c] * rsqrtf(va3[c] + 1e-5f); SC[j] = (b3[c] - m3[c]) * s + be3[c]; }
      else              { int c = j - 914; SC[j] = b4[c]; }
    }
  }
}

// ---------------- init: pass through unq, pooled := -inf ----------------
__global__ void init_kernel(const long long* __restrict__ unq,
                            long long* __restrict__ out_unq,
                            float* __restrict__ pooled, int U)
{
  int i = blockIdx.x * blockDim.x + threadIdx.x;
  if (i < U * 4)   out_unq[i] = unq[i];
  if (i < U * 256) pooled[i] = __int_as_float(0xFF800000u);  // -inf
}

// ---------------- fragment helpers (bf16 WMMA, wave32) ----------------
// A 16x32 bf16: lanes 0-15 -> M=lane, K 0..7 & 16..23 ; lanes 16-31 -> M=lane-16, K 8..15 & 24..31
__device__ __forceinline__ v16bf ld_a(const __bf16* act, int lane, int kt, int ldk) {
  int r  = lane & 15;
  int kb = (kt << 5) + ((lane >> 4) << 3);
  v8bf lo = *(const v8bf*)(act + r * ldk + kb);
  v8bf hi = *(const v8bf*)(act + r * ldk + kb + 16);
  return __builtin_shufflevector(lo, hi, 0,1,2,3,4,5,6,7,8,9,10,11,12,13,14,15);
}
// B 32x16 bf16 from transposed weights wt[out][K]: lane -> column, K 0..7 & 16..23 (lanes 0-15) / 8..15 & 24..31 (lanes 16-31)
__device__ __forceinline__ v16bf ld_b(const __bf16* wt, int lane, int nt, int kt, int K) {
  int c  = (lane & 15) + (nt << 4);
  int kb = (kt << 5) + ((lane >> 4) << 3);
  v8bf lo = *(const v8bf*)(wt + c * K + kb);
  v8bf hi = *(const v8bf*)(wt + c * K + kb + 16);
  return __builtin_shufflevector(lo, hi, 0,1,2,3,4,5,6,7,8,9,10,11,12,13,14,15);
}
__device__ __forceinline__ v8f bfwmma(v16bf a, v16bf b, v8f c) {
  return __builtin_amdgcn_wmma_f32_16x16x32_bf16(false, a, false, b, (short)0, c, false, false);
}
// C/D 16x16 f32: VGPR j, lanes 0-15 -> (M=j, N=lane), lanes 16-31 -> (M=j+8, N=lane-16)
__device__ __forceinline__ void st_act(__bf16* act, int lane, int nt, v8f c,
                                       const float* s, const float* t, int ldn) {
  int col = (lane & 15) + (nt << 4);
  int r0  = (lane >> 4) << 3;
  float sc = s[col], tb = t[col];
#pragma unroll
  for (int j = 0; j < 8; j++) {
    float v = fmaxf(fmaf(c[j], sc, tb), 0.f);  // BN + ReLU
    act[(r0 + j) * ldn + col] = (__bf16)v;
  }
}
__device__ __forceinline__ void atomicMaxF(float* a, float v) {
  if (!(v < 0.f)) atomicMax((int*)a, __float_as_int(v));
  else            atomicMin((unsigned int*)a, __float_as_uint(v));
}

// ---------------- main fused MLP + segment-max kernel ----------------
#define WAVES 4
__global__ void __launch_bounds__(128, 1)
mlp_pool_kernel(const float* __restrict__ x, const int* __restrict__ unq_inv,
                const char* __restrict__ ws, float* __restrict__ pooled, int Nrows)
{
  __shared__ __align__(32) __bf16 sW[43008];            // w1t(2048)+w2t(8192)+w3t(32768) bf16
  __shared__ __align__(16) float  sSC[1176];            // folded BN scale/bias table
  __shared__ __align__(32) __bf16 sBuf0[WAVES][16 * 128];
  __shared__ __align__(32) __bf16 sBuf1[WAVES][16 * 256];
  __shared__ int sSeg[WAVES][16];

  const int tid = threadIdx.x, lane = tid & 31, wv = tid >> 5;

  // stage w1/w2/w3 (86016 B = 5376 x b128) and scales (293 x b128) into LDS
#if USE_ASYNC_LDS
  {
    g_v4i* gw = (g_v4i*)(char*)ws;                       // workspace base, AS1, 16B units
    g_v4i* gs = (g_v4i*)((char*)ws + OFF_SC);
    l_v4i* lw = (l_v4i*)sW;
    l_v4i* ls = (l_v4i*)sSC;
    for (int i = tid; i < 5376; i += 128)
      __builtin_amdgcn_global_load_async_to_lds_b128(gw + i, lw + i, 0, 0);
    for (int i = tid; i < 293; i += 128)
      __builtin_amdgcn_global_load_async_to_lds_b128(gs + i, ls + i, 0, 0);
    __builtin_amdgcn_s_wait_asynccnt(0);
  }
#else
  {
    const uint4* src = (const uint4*)ws;
    uint4* dst = (uint4*)sW;
    for (int i = tid; i < 5376; i += 128) dst[i] = src[i];
    const uint4* src2 = (const uint4*)(ws + OFF_SC);
    uint4* dst2 = (uint4*)sSC;
    for (int i = tid; i < 293; i += 128) dst2[i] = src2[i];
  }
#endif
  const float *s0 = sSC,       *t0 = sSC + 9,   *s1 = sSC + 18,  *t1 = sSC + 82;
  const float *s2 = sSC + 146, *t2 = sSC + 274, *s3 = sSC + 402, *t3 = sSC + 658;
  const float *b4 = sSC + 914;
  const __bf16* sW1 = sW;
  const __bf16* sW2 = sW + 2048;
  const __bf16* sW3 = sW + 10240;
  const __bf16* gW4 = (const __bf16*)(ws + OFF_W4T);

  const int rowBase = blockIdx.x * (WAVES * 16) + wv * 16;
  __bf16* A0 = &sBuf0[wv][0];
  __bf16* A1 = &sBuf1[wv][0];

  // stage raw input rows + segment ids (BN0 applied after barrier, scales are in LDS)
  float xr[9];
  {
    int r = lane & 15;
    int row = rowBase + r; if (row >= Nrows) row = Nrows - 1;
    if (lane < 16) {
      const float* xp = x + (long)row * 9;
#pragma unroll
      for (int k = 0; k < 9; k++) xr[k] = xp[k];
      sSeg[wv][r] = unq_inv[row];
    } else {
#pragma unroll
      for (int k = 9; k < 32; k++) A0[r * 32 + k] = (__bf16)0.f;
    }
  }
  __syncthreads();
  {
    int r = lane & 15;
    if (lane < 16) {
#pragma unroll
      for (int k = 0; k < 9; k++) A0[r * 32 + k] = (__bf16)fmaf(xr[k], s0[k], t0[k]);
    }
  }

  // ---- L1: 9(pad 32) -> 64, A0 -> A1 ----
  {
    v16bf a = ld_a(A0, lane, 0, 32);
    v8f c4[4];
#pragma unroll
    for (int nt = 0; nt < 4; nt++) {
      v8f z = {};
      c4[nt] = bfwmma(a, ld_b(sW1, lane, nt, 0, 32), z);
    }
#pragma unroll
    for (int nt = 0; nt < 4; nt++) st_act(A1, lane, nt, c4[nt], s1, t1, 64);
  }

  // ---- L2: 64 -> 128, A1 -> A0 ----
  {
    v8f c8[8];
#pragma unroll
    for (int nt = 0; nt < 8; nt++) c8[nt] = (v8f){};
#pragma unroll
    for (int kt = 0; kt < 2; kt++) {
      v16bf a = ld_a(A1, lane, kt, 64);
#pragma unroll
      for (int nt = 0; nt < 8; nt++) c8[nt] = bfwmma(a, ld_b(sW2, lane, nt, kt, 64), c8[nt]);
    }
#pragma unroll
    for (int nt = 0; nt < 8; nt++) st_act(A0, lane, nt, c8[nt], s2, t2, 128);
  }

  // ---- L3: 128 -> 256, A0 -> A1 (two N-blocks to cap VGPRs) ----
#pragma unroll
  for (int nb = 0; nb < 2; nb++) {
    v8f acc[8];
#pragma unroll
    for (int nt = 0; nt < 8; nt++) acc[nt] = (v8f){};
#pragma unroll
    for (int kt = 0; kt < 4; kt++) {
      v16bf a = ld_a(A0, lane, kt, 128);
#pragma unroll
      for (int nt = 0; nt < 8; nt++)
        acc[nt] = bfwmma(a, ld_b(sW3, lane, nb * 8 + nt, kt, 128), acc[nt]);
    }
#pragma unroll
    for (int nt = 0; nt < 8; nt++) st_act(A1, lane, nb * 8 + nt, acc[nt], s3, t3, 256);
  }

  // ---- L4: 256 -> 256 (w4 streamed from L2-resident ws), bias init, segment-max atomics ----
#pragma unroll
  for (int nb = 0; nb < 2; nb++) {
    v8f acc[8];
#pragma unroll
    for (int nt = 0; nt < 8; nt++) {
      float bv = b4[(lane & 15) + ((nb * 8 + nt) << 4)];
      acc[nt] = (v8f){bv, bv, bv, bv, bv, bv, bv, bv};
    }
#pragma unroll
    for (int kt = 0; kt < 8; kt++) {
      v16bf a = ld_a(A1, lane, kt, 256);
      if (kt < 7) __builtin_prefetch(gW4 + ((lane & 15) + (nb << 7)) * 256 + ((kt + 1) << 5), 0, 1);
#pragma unroll
      for (int nt = 0; nt < 8; nt++)
        acc[nt] = bfwmma(a, ld_b(gW4, lane, nb * 8 + nt, kt, 256), acc[nt]);
    }
    int r0 = (lane >> 4) << 3;
#pragma unroll
    for (int nt = 0; nt < 8; nt++) {
      int col = (lane & 15) + ((nb * 8 + nt) << 4);
#pragma unroll
      for (int j = 0; j < 8; j++) {
        int row = rowBase + r0 + j;
        if (row < Nrows) {
          long seg = (long)sSeg[wv][r0 + j];
          atomicMaxF(pooled + seg * 256 + col, acc[nt][j]);
        }
      }
    }
  }
}

// ---------------- host entry ----------------
extern "C" void kernel_launch(void* const* d_in, const int* in_sizes, int n_in,
                              void* d_out, int out_size, void* d_ws, size_t ws_size,
                              hipStream_t stream) {
  const float*     x       = (const float*)d_in[0];
  const long long* unq     = (const long long*)d_in[1];
  const int*       unq_inv = (const int*)d_in[2];
  const float* bn[16];
  for (int i = 0; i < 16; i++) bn[i] = (const float*)d_in[3 + i];
  const float* w1 = (const float*)d_in[19]; const float* b1 = (const float*)d_in[20];
  const float* w2 = (const float*)d_in[21]; const float* b2 = (const float*)d_in[22];
  const float* w3 = (const float*)d_in[23]; const float* b3 = (const float*)d_in[24];
  const float* w4 = (const float*)d_in[25]; const float* b4 = (const float*)d_in[26];

  int N = in_sizes[2];
  int U = in_sizes[1] / 4;
  char* ws = (char*)d_ws;
  long long* out_unq = (long long*)d_out;
  float* pooled = (float*)d_out + (size_t)U * 8;  // after U*4 int64 entries

  prep_kernel<<<(108544 + 1170 + 255) / 256, 256, 0, stream>>>(
      bn[0], bn[1], bn[2], bn[3], bn[4], bn[5], bn[6], bn[7],
      bn[8], bn[9], bn[10], bn[11], bn[12], bn[13], bn[14], bn[15],
      w1, b1, w2, b2, w3, b3, w4, b4, ws);

  if (U > 0)
    init_kernel<<<((size_t)U * 256 + 255) / 256, 256, 0, stream>>>(unq, out_unq, pooled, U);

  if (N > 0) {
    int blocks = (N + WAVES * 16 - 1) / (WAVES * 16);
    mlp_pool_kernel<<<blocks, 128, 0, stream>>>(x, unq_inv, ws, pooled, N);
  }
}
